// DualTransformerBlock_10960756540214
// MI455X (gfx1250) — matrix operand order repaired and checked by reading, not compiled, not run on hardware
//
#include <hip/hip_runtime.h>
#include <math.h>

// ---------------------------------------------------------------------------
// Problem constants (from reference)
// ---------------------------------------------------------------------------
constexpr int B_   = 4;
constexpr int N_   = 4096;
constexpr int C_   = 256;
constexpr int BN_  = B_ * N_;        // 16384 rows
constexpr int H_   = 8;              // channel-attention heads
constexpr int HD_  = C_ / H_;        // 32
constexpr int DFF_ = 4 * C_;         // 1024

// ---------------------------------------------------------------------------
// WMMA types (gfx1250, wave32)
// ---------------------------------------------------------------------------
typedef __attribute__((ext_vector_type(16))) _Float16 v16h;
typedef __attribute__((ext_vector_type(8)))  _Float16 v8h;
typedef __attribute__((ext_vector_type(4)))  _Float16 v4h;
typedef __attribute__((ext_vector_type(8)))  float    v8f;

static __device__ __forceinline__ v16h cat8(v8h lo, v8h hi) {
    return __builtin_shufflevector(lo, hi, 0, 1, 2, 3, 4, 5, 6, 7,
                                           8, 9, 10, 11, 12, 13, 14, 15);
}

// ---------------------------------------------------------------------------
// Fused WMMA GEMM:  Y = act( A_eff @ B_eff + bias ) + residual
//   A_eff[m][k] = TA ? A[k*lda + m] : A[m*lda + k]
//   B_eff[k][n] = TB ? Bw[n*ldb + k] : Bw[k*ldb + n]      (TB=1 == X @ W^T)
// fp32 in/out, f16 WMMA, f32 accumulate.
// Block 256 thr = 8 waves; tile 128x128x32; wave tile 32x64 => 2x4 wmma/step.
// Double-buffered LDS: global loads for tile k+1 register-staged while tile k
// computes; single barrier per K step. A tile m-major [m][k], B tile n-major
// [n][k] so every fragment is two 16B-aligned ds_load_b128 + one shuffle.
// ---------------------------------------------------------------------------
template <int TA, int TB>
__launch_bounds__(256)
__global__ void gemm_wmma_f16(const float* __restrict__ A, long long sAb, int lda,
                              const float* __restrict__ Bw, long long sBb, int ldb,
                              const float* __restrict__ bias,
                              const float* __restrict__ res, long long sRb, int ldr,
                              float* __restrict__ Y, long long sYb, int ldy,
                              int M, int Nn, int K, int act)
{
    constexpr int BM = 128, TBN = 128, BK = 32;
    constexpr int LR = BK + 8;                 // 40 halves -> 80B row (16B mult)
    __shared__ _Float16 As[2][BM][LR];
    __shared__ _Float16 Bs[2][TBN][LR];

    const int tid  = threadIdx.x;
    const int lane = tid & 31;
    const int w    = tid >> 5;                 // wave 0..7
    const int wm   = (w >> 1) * 32;            // 4 waves along M
    const int wn   = (w & 1) * 64;             // 2 waves along N
    const int mt   = blockIdx.y * BM;
    const int nt   = blockIdx.x * TBN;

    const float* Ab = A  + (long long)blockIdx.z * sAb;
    const float* Bb = Bw + (long long)blockIdx.z * sBb;
    float*       Yb = Y  + (long long)blockIdx.z * sYb;
    const float* Rb = res ? res + (long long)blockIdx.z * sRb : nullptr;

    v8f acc[2][4] = {};
    const int l16 = lane & 15;
    const int hf  = lane >> 4;

    // ---- register-staged tile loaders / LDS stores ----
    auto loadA = [&](int k0, float4* fa) {
        #pragma unroll
        for (int it = 0; it < 4; ++it) {
            int e4 = tid + it * 256;
            if constexpr (TA == 0) {
                int i = e4 >> 3, k4 = (e4 & 7) * 4;
                fa[it] = *(const float4*)&Ab[(long long)(mt + i) * lda + (k0 + k4)];
            } else {
                int k = e4 >> 5, i4 = (e4 & 31) * 4;
                fa[it] = *(const float4*)&Ab[(long long)(k0 + k) * lda + (mt + i4)];
            }
        }
    };
    auto loadB = [&](int k0, float4* fb) {
        #pragma unroll
        for (int it = 0; it < 4; ++it) {
            int e4 = tid + it * 256;
            if constexpr (TB == 1) {
                int n = e4 >> 3, k4 = (e4 & 7) * 4;
                fb[it] = *(const float4*)&Bb[(long long)(nt + n) * ldb + (k0 + k4)];
            } else {
                int k = e4 >> 5, n4 = (e4 & 31) * 4;
                fb[it] = *(const float4*)&Bb[(long long)(k0 + k) * ldb + (nt + n4)];
            }
        }
    };
    auto storeA = [&](int buf, const float4* fa) {
        #pragma unroll
        for (int it = 0; it < 4; ++it) {
            int e4 = tid + it * 256;
            if constexpr (TA == 0) {
                int i = e4 >> 3, k4 = (e4 & 7) * 4;
                v4h h = { (_Float16)fa[it].x, (_Float16)fa[it].y,
                          (_Float16)fa[it].z, (_Float16)fa[it].w };
                *(v4h*)&As[buf][i][k4] = h;
            } else {
                int k = e4 >> 5, i4 = (e4 & 31) * 4;
                As[buf][i4 + 0][k] = (_Float16)fa[it].x;
                As[buf][i4 + 1][k] = (_Float16)fa[it].y;
                As[buf][i4 + 2][k] = (_Float16)fa[it].z;
                As[buf][i4 + 3][k] = (_Float16)fa[it].w;
            }
        }
    };
    auto storeB = [&](int buf, const float4* fb) {
        #pragma unroll
        for (int it = 0; it < 4; ++it) {
            int e4 = tid + it * 256;
            if constexpr (TB == 1) {
                int n = e4 >> 3, k4 = (e4 & 7) * 4;
                v4h h = { (_Float16)fb[it].x, (_Float16)fb[it].y,
                          (_Float16)fb[it].z, (_Float16)fb[it].w };
                *(v4h*)&Bs[buf][n][k4] = h;
            } else {
                int k = e4 >> 5, n4 = (e4 & 31) * 4;
                Bs[buf][n4 + 0][k] = (_Float16)fb[it].x;
                Bs[buf][n4 + 1][k] = (_Float16)fb[it].y;
                Bs[buf][n4 + 2][k] = (_Float16)fb[it].z;
                Bs[buf][n4 + 3][k] = (_Float16)fb[it].w;
            }
        }
    };

    // ---- prologue: stage tile 0 ----
    float4 fa[4], fb[4];
    loadA(0, fa); loadB(0, fb);
    storeA(0, fa); storeB(0, fb);
    __syncthreads();

    int cur = 0;
    for (int k0 = 0; k0 < K; k0 += BK) {
        const bool more = (k0 + BK) < K;
        // issue next tile's global loads (latency hidden under WMMA)
        if (more) { loadA(k0 + BK, fa); loadB(k0 + BK, fb); }
        // prefetch the tile after next (global_prefetch_b8)
        if (k0 + 2 * BK < K) {
            const float* pa = TA ? Ab + (long long)(k0 + 2 * BK) * lda + mt + (tid >> 1)
                                 : Ab + (long long)(mt + (tid >> 1)) * lda + (k0 + 2 * BK);
            __builtin_prefetch(pa, 0, 0);
            const float* pb = TB ? Bb + (long long)(nt + (tid >> 1)) * ldb + (k0 + 2 * BK)
                                 : Bb + (long long)(k0 + 2 * BK) * ldb + nt + (tid >> 1);
            __builtin_prefetch(pb, 0, 0);
        }

        // ---- fragments: contiguous 16B LDS loads (ISA 7.12.2 layouts) ----
        v16h afr[2], bfr[4];
        #pragma unroll
        for (int t = 0; t < 2; ++t) {
            const _Float16* pr = &As[cur][wm + t * 16 + l16][hf * 8];
            afr[t] = cat8(*(const v8h*)pr, *(const v8h*)(pr + 16));
        }
        #pragma unroll
        for (int t = 0; t < 4; ++t) {
            const _Float16* pb = &Bs[cur][wn + t * 16 + l16][hf * 8];
            bfr[t] = cat8(*(const v8h*)pb, *(const v8h*)(pb + 16));
        }

        // ---- 8x v_wmma_f32_16x16x32_f16 per K step ----
        #pragma unroll
        for (int tm = 0; tm < 2; ++tm)
            #pragma unroll
            for (int tn = 0; tn < 4; ++tn)
                acc[tm][tn] = __builtin_amdgcn_wmma_f32_16x16x32_f16(
                    false, afr[tm], false, bfr[tn], (short)0, acc[tm][tn], false, false);

        // convert + store next tile into the idle buffer
        if (more) { storeA(cur ^ 1, fa); storeB(cur ^ 1, fb); }
        __syncthreads();
        cur ^= 1;
    }

    // ---- epilogue: bias + exact GELU + residual ----
    #pragma unroll
    for (int tm = 0; tm < 2; ++tm)
        #pragma unroll
        for (int tn = 0; tn < 4; ++tn)
            #pragma unroll
            for (int r = 0; r < 8; ++r) {
                int gm = mt + wm + tm * 16 + hf * 8 + r;
                int gn = nt + wn + tn * 16 + l16;
                float v = acc[tm][tn][r];
                if (bias) v += bias[gn];
                if (act)  v = 0.5f * v * (1.0f + erff(v * 0.70710678118654752f));
                if (Rb)   v += Rb[(long long)gm * ldr + gn];
                Yb[(long long)gm * ldy + gn] = v;
            }
}

// ---------------------------------------------------------------------------
// LayerNorm over last dim (C_=256), one block (256 thr) per row
// ---------------------------------------------------------------------------
__launch_bounds__(256)
__global__ void ln_kernel(const float* __restrict__ x, const float* __restrict__ g,
                          const float* __restrict__ b, float* __restrict__ y)
{
    __shared__ float s[256];
    const int row = blockIdx.x, t = threadIdx.x;
    const float v = x[(long long)row * C_ + t];
    s[t] = v; __syncthreads();
    for (int o = 128; o > 0; o >>= 1) { if (t < o) s[t] += s[t + o]; __syncthreads(); }
    const float mu = s[0] * (1.0f / C_); __syncthreads();
    const float d = v - mu;
    s[t] = d * d; __syncthreads();
    for (int o = 128; o > 0; o >>= 1) { if (t < o) s[t] += s[t + o]; __syncthreads(); }
    const float var = s[0] * (1.0f / C_);
    y[(long long)row * C_ + t] = d * rsqrtf(var + 1e-5f) * g[t] + b[t];
}

// ---------------------------------------------------------------------------
// Softmax over channel dim (axis=2): one block per row, in place
// ---------------------------------------------------------------------------
__launch_bounds__(256)
__global__ void row_softmax(float* __restrict__ x)
{
    __shared__ float s[256];
    const int row = blockIdx.x, t = threadIdx.x;
    const long long idx = (long long)row * C_ + t;
    const float v = x[idx];
    s[t] = v; __syncthreads();
    for (int o = 128; o > 0; o >>= 1) { if (t < o) s[t] = fmaxf(s[t], s[t + o]); __syncthreads(); }
    const float mx = s[0]; __syncthreads();
    const float e = __expf(v - mx);
    s[t] = e; __syncthreads();
    for (int o = 128; o > 0; o >>= 1) { if (t < o) s[t] += s[t + o]; __syncthreads(); }
    x[idx] = e / s[0];
}

// ---------------------------------------------------------------------------
// Softmax over token dim (axis=1): one block per (b, channel), stride C_
// ---------------------------------------------------------------------------
__launch_bounds__(256)
__global__ void col_softmax(float* __restrict__ x)
{
    __shared__ float s[256];
    const int col = blockIdx.x;             // 0..B*C-1
    const int b = col / C_, c = col % C_;
    const long long base = (long long)b * N_ * C_ + c;
    const int t = threadIdx.x;
    float loc[16];
    float mx = -3.0e38f;
    #pragma unroll
    for (int i = 0; i < 16; ++i) {
        float v = x[base + (long long)(t + i * 256) * C_];
        loc[i] = v; mx = fmaxf(mx, v);
    }
    s[t] = mx; __syncthreads();
    for (int o = 128; o > 0; o >>= 1) { if (t < o) s[t] = fmaxf(s[t], s[t + o]); __syncthreads(); }
    mx = s[0]; __syncthreads();
    float sum = 0.0f;
    #pragma unroll
    for (int i = 0; i < 16; ++i) { loc[i] = __expf(loc[i] - mx); sum += loc[i]; }
    s[t] = sum; __syncthreads();
    for (int o = 128; o > 0; o >>= 1) { if (t < o) s[t] += s[t + o]; __syncthreads(); }
    const float inv = 1.0f / s[0];
    #pragma unroll
    for (int i = 0; i < 16; ++i) x[base + (long long)(t + i * 256) * C_] = loc[i] * inv;
}

// ---------------------------------------------------------------------------
// Channel-attention q/k L2-normalize over N: one block per (part, b, channel)
//   qkv rows are [q(256) | k(256) | v(256)]; output layout [B, C, N]
// ---------------------------------------------------------------------------
__launch_bounds__(256)
__global__ void qk_normalize(const float* __restrict__ qkv,
                             float* __restrict__ qn, float* __restrict__ kn)
{
    __shared__ float s[256];
    const int id = blockIdx.x;              // 0..2*B*C-1
    const int part = id / (B_ * C_);
    const int rem  = id % (B_ * C_);
    const int b = rem / C_, ch = rem % C_;
    const float* src = qkv + (long long)b * N_ * 3 * C_ + part * C_ + ch;
    float* dst = (part == 0 ? qn : kn) + ((long long)b * C_ + ch) * N_;
    const int t = threadIdx.x;
    float loc[16]; float ss = 0.0f;
    #pragma unroll
    for (int i = 0; i < 16; ++i) {
        float v = src[(long long)(t + i * 256) * 3 * C_];
        loc[i] = v; ss += v * v;
    }
    s[t] = ss; __syncthreads();
    for (int o = 128; o > 0; o >>= 1) { if (t < o) s[t] += s[t + o]; __syncthreads(); }
    const float inv = 1.0f / fmaxf(sqrtf(s[0]), 1e-12f);
    #pragma unroll
    for (int i = 0; i < 16; ++i) dst[t + i * 256] = loc[i] * inv;
}

// ---------------------------------------------------------------------------
// Channel-attention core: per (b,h): attn = softmax((q̂ k̂ᵀ)·temp), out = attn·v
//   qn/kn: [B, C, N]; v from qkv rows (offset 2C); out written as [B,N,C]
// ---------------------------------------------------------------------------
__launch_bounds__(1024)
__global__ void chan_attn_core(const float* __restrict__ qn, const float* __restrict__ kn,
                               const float* __restrict__ qkv, const float* __restrict__ temp,
                               float* __restrict__ outbuf)
{
    __shared__ float attn[HD_][HD_ + 1];
    const int bh = blockIdx.x;
    const int b = bh / H_, h = bh % H_;
    const int t = threadIdx.x;
    const int c = t >> 5, d = t & 31;

    // attn[c][d] = temp[h] * sum_n q̂[c,n] k̂[d,n]  (float4 vectorized)
    const float4* q4 = (const float4*)(qn + ((long long)b * C_ + h * HD_ + c) * N_);
    const float4* k4 = (const float4*)(kn + ((long long)b * C_ + h * HD_ + d) * N_);
    float sum = 0.0f;
    for (int n = 0; n < N_ / 4; ++n) {
        float4 a = q4[n], bb = k4[n];
        sum += a.x * bb.x + a.y * bb.y + a.z * bb.z + a.w * bb.w;
    }
    attn[c][d] = sum * temp[h];
    __syncthreads();

    // per-row softmax: wave handles row c (=wave id), lane = column d
    {
        const float v = attn[c][d];
        float m = v;
        #pragma unroll
        for (int o = 16; o > 0; o >>= 1) m = fmaxf(m, __shfl_xor(m, o, 32));
        const float e = __expf(v - m);
        float s2 = e;
        #pragma unroll
        for (int o = 16; o > 0; o >>= 1) s2 += __shfl_xor(s2, o, 32);
        attn[c][d] = e / s2;
    }
    __syncthreads();

    // out[b,n,h*HD+c] = sum_d attn[c][d] * v[b,h,d,n]
    for (int idx = t; idx < HD_ * N_; idx += 1024) {
        const int n  = idx >> 5;
        const int c2 = idx & 31;
        const float4* v4 = (const float4*)(qkv + ((long long)(b * N_ + n)) * 3 * C_ + 2 * C_ + h * HD_);
        float acc = 0.0f;
        #pragma unroll
        for (int dd = 0; dd < HD_ / 4; ++dd) {
            float4 vv = v4[dd];
            acc += attn[c2][4 * dd + 0] * vv.x + attn[c2][4 * dd + 1] * vv.y
                 + attn[c2][4 * dd + 2] * vv.z + attn[c2][4 * dd + 3] * vv.w;
        }
        outbuf[((long long)(b * N_ + n)) * C_ + h * HD_ + c2] = acc;
    }
}

// ---------------------------------------------------------------------------
// Host orchestration
// ---------------------------------------------------------------------------
extern "C" void kernel_launch(void* const* d_in, const int* in_sizes, int n_in,
                              void* d_out, int out_size, void* d_ws, size_t ws_size,
                              hipStream_t stream)
{
    (void)in_sizes; (void)n_in; (void)out_size;

    const float* x      = (const float*)d_in[0];
    const float* ln1_g  = (const float*)d_in[1];
    const float* ln1_b  = (const float*)d_in[2];
    const float* ln2_g  = (const float*)d_in[3];
    const float* ln2_b  = (const float*)d_in[4];
    const float* ln3_g  = (const float*)d_in[5];
    const float* ln3_b  = (const float*)d_in[6];
    const float* ln4_g  = (const float*)d_in[7];
    const float* ln4_b  = (const float*)d_in[8];
    const float* wk     = (const float*)d_in[9];
    const float* wq     = (const float*)d_in[10];
    const float* wv     = (const float*)d_in[11];
    const float* wr     = (const float*)d_in[12];
    const float* qkv_w  = (const float*)d_in[13];
    const float* temper = (const float*)d_in[14];
    const float* proj_w = (const float*)d_in[15];
    const float* proj_b = (const float*)d_in[16];
    const float* m1_w1  = (const float*)d_in[17];
    const float* m1_b1  = (const float*)d_in[18];
    const float* m1_w2  = (const float*)d_in[19];
    const float* m1_b2  = (const float*)d_in[20];
    const float* m2_w1  = (const float*)d_in[21];
    const float* m2_b1  = (const float*)d_in[22];
    const float* m2_w2  = (const float*)d_in[23];
    const float* m2_b2  = (const float*)d_in[24];
    float* out = (float*)d_out;

    const long long A = (long long)BN_ * C_;           // 4,194,304 floats
    const size_t need = (size_t)(11 * A + (long long)B_ * C_ * C_) * sizeof(float);
    if (ws_size < need) return;

    float* ws   = (float*)d_ws;
    float* nbuf = ws;               // [BN,C]   LN outputs / qn / add3
    float* qkv  = ws + A;           // [BN,3C]  also K|Q|V in phase 1
    float* keys = qkv;              // [BN,C]
    float* qrs  = qkv + A;          // [BN,C]
    float* vals = qkv + 2 * A;      // [BN,C]
    float* tmp  = ws + 4 * A;       // [BN,C]   att / kn / n4
    float* r1   = ws + 5 * A;       // [BN,C]   add1 / ctx_out
    float* r2   = ws + 6 * A;       // [BN,C]   add2
    float* hid  = ws + 7 * A;       // [BN,DFF]
    float* ktq  = ws + 11 * A;      // [B,C,C]

    const dim3 blk256(256);

    auto gemm = [&](int TA, int TB,
                    const float* Am, long long sAb, int lda,
                    const float* Bm, long long sBb, int ldb,
                    const float* bias, const float* res, long long sRb, int ldr,
                    float* Ym, long long sYb, int ldy,
                    int M, int Nn, int K, int act, int batch) {
        dim3 grid(Nn / 128, M / 128, batch);
        if (TA == 0 && TB == 1)
            gemm_wmma_f16<0, 1><<<grid, blk256, 0, stream>>>(
                Am, sAb, lda, Bm, sBb, ldb, bias, res, sRb, ldr, Ym, sYb, ldy, M, Nn, K, act);
        else if (TA == 1 && TB == 0)
            gemm_wmma_f16<1, 0><<<grid, blk256, 0, stream>>>(
                Am, sAb, lda, Bm, sBb, ldb, bias, res, sRb, ldr, Ym, sYb, ldy, M, Nn, K, act);
        else
            gemm_wmma_f16<0, 0><<<grid, blk256, 0, stream>>>(
                Am, sAb, lda, Bm, sBb, ldb, bias, res, sRb, ldr, Ym, sYb, ldy, M, Nn, K, act);
    };

    const long long sNC = (long long)N_ * C_;
    const long long sCC = (long long)C_ * C_;

    // ---- EfficientAttention branch ----
    ln_kernel<<<BN_, 256, 0, stream>>>(x, ln1_g, ln1_b, nbuf);                        // n1
    gemm(0, 1, nbuf, 0, C_, wk, 0, C_, nullptr, nullptr, 0, 0, keys, 0, C_, BN_, C_, C_, 0, 1);
    gemm(0, 1, nbuf, 0, C_, wq, 0, C_, nullptr, nullptr, 0, 0, qrs,  0, C_, BN_, C_, C_, 0, 1);
    gemm(0, 1, nbuf, 0, C_, wv, 0, C_, nullptr, nullptr, 0, 0, vals, 0, C_, BN_, C_, C_, 0, 1);
    row_softmax<<<BN_, 256, 0, stream>>>(keys);                                       // softmax axis=2
    col_softmax<<<B_ * C_, 256, 0, stream>>>(qrs);                                    // softmax axis=1
    // ktq[b] = k[b]^T @ q[b]   (M=C, N=C, K=N_)
    gemm(1, 0, keys, sNC, C_, qrs, sNC, C_, nullptr, nullptr, 0, 0, ktq, sCC, C_, C_, C_, N_, 0, B_);
    // att[b] = v[b] @ ktq[b]   (M=N_, N=C, K=C)
    gemm(0, 0, vals, sNC, C_, ktq, sCC, C_, nullptr, nullptr, 0, 0, tmp, sNC, C_, N_, C_, C_, 0, B_);
    // add1 = x + att @ wr^T
    gemm(0, 1, tmp, 0, C_, wr, 0, C_, nullptr, x, 0, C_, r1, 0, C_, BN_, C_, C_, 0, 1);

    // ---- MLP 1 ----
    ln_kernel<<<BN_, 256, 0, stream>>>(r1, ln2_g, ln2_b, nbuf);                       // n2
    gemm(0, 1, nbuf, 0, C_, m1_w1, 0, C_, m1_b1, nullptr, 0, 0, hid, 0, DFF_, BN_, DFF_, C_, 1, 1);
    gemm(0, 1, hid, 0, DFF_, m1_w2, 0, DFF_, m1_b2, r1, 0, C_, r2, 0, C_, BN_, C_, DFF_, 0, 1);

    // ---- ChannelAttention branch ----
    ln_kernel<<<BN_, 256, 0, stream>>>(r2, ln3_g, ln3_b, nbuf);                       // n3
    gemm(0, 1, nbuf, 0, C_, qkv_w, 0, C_, nullptr, nullptr, 0, 0, qkv, 0, 3 * C_, BN_, 3 * C_, C_, 0, 1);
    qk_normalize<<<2 * B_ * C_, 256, 0, stream>>>(qkv, nbuf, tmp);                    // qn->nbuf, kn->tmp
    chan_attn_core<<<B_ * H_, 1024, 0, stream>>>(nbuf, tmp, qkv, temper, r1);         // ctx -> r1
    // add3 = add2 + ctx @ proj_w^T + proj_b
    gemm(0, 1, r1, 0, C_, proj_w, 0, C_, proj_b, r2, 0, C_, nbuf, 0, C_, BN_, C_, C_, 0, 1);

    // ---- MLP 2 ----
    ln_kernel<<<BN_, 256, 0, stream>>>(nbuf, ln4_g, ln4_b, tmp);                      // n4
    gemm(0, 1, tmp, 0, C_, m2_w1, 0, C_, m2_b1, nullptr, 0, 0, hid, 0, DFF_, BN_, DFF_, C_, 1, 1);
    gemm(0, 1, hid, 0, DFF_, m2_w2, 0, DFF_, m2_b2, nbuf, 0, C_, out, 0, C_, BN_, C_, DFF_, 0, 1);
}
